// MultiHeadAttention_20718922236412
// MI455X (gfx1250) — matrix operand order repaired and checked by reading, not compiled, not run on hardware
//
#include <hip/hip_runtime.h>

// ---------------------------------------------------------------------------
// MHA for MI455X (gfx1250, wave32). All GEMMs via v_wmma_f32_16x16x32_f16.
// B=16, S=1024, D=512, H=8, DK=DV=64.
// Round 4: fully unroll pipelined GEMM K-loops (kills v_dual_mov rotation
// copies, keeps partial s_wait_loadcnt); unroll-by-2 attention key loop.
// ---------------------------------------------------------------------------

typedef __attribute__((ext_vector_type(16))) _Float16 v16h;
typedef __attribute__((ext_vector_type(8)))  _Float16 v8h;
typedef __attribute__((ext_vector_type(8)))  float    v8f;

#define NB   16
#define NS   1024
#define ND   512
#define NH   8
#define NDK  64

static __device__ __forceinline__ v8f wmma16(v16h a, v16h b, v8f c) {
  // v_wmma_f32_16x16x32_f16  (8-arg probe-confirmed signature)
  return __builtin_amdgcn_wmma_f32_16x16x32_f16(false, a, false, b, (short)0, c,
                                                false, false);
}

// A-fragment (16x32 f16) from a row-major matrix, per-lane:
//   elems 0..15 -> K = k0 + hf*16 + j       (contiguous 32B)
static __device__ __forceinline__ v16h load_a_frag(const _Float16* base, int row,
                                                   int stride, int k0, int hf) {
  const _Float16* p = base + (size_t)row * stride + k0 + hf * 16;
  v16h a;
  *(v8h*)&a       = *(const v8h*)p;
  *((v8h*)&a + 1) = *(const v8h*)(p + 8);
  return a;
}

// 16 contiguous halves -> one fragment register group (two b128 loads)
static __device__ __forceinline__ v16h load_frag16(const _Float16* p) {
  v16h r;
  *(v8h*)&r       = *(const v8h*)p;
  *((v8h*)&r + 1) = *(const v8h*)(p + 8);
  return r;
}

// ---------------------------------------------------------------------------
// CDNA5 async global->LDS copy (ASYNCcnt-tracked), guarded by __has_builtin.
// Builtin signature (from hipcc diagnostic): (global v4i*, lds v4i*, off, cpol).
// AS1 ptr = 64-bit global addr; AS3 ptr = LDS byte offset = low 32 bits of the
// generic shared address (ISA 10.2).
// ---------------------------------------------------------------------------
#if __has_builtin(__builtin_amdgcn_global_load_async_to_lds_b128)
#define USE_ASYNC_LDS 1
typedef int v4i_vs __attribute__((vector_size(4 * sizeof(int))));
typedef __attribute__((address_space(1))) v4i_vs* as1_v4i;
typedef __attribute__((address_space(3))) v4i_vs* as3_v4i;
static __device__ __forceinline__ void async_copy16(_Float16* lds_dst,
                                                    const _Float16* gsrc) {
  __builtin_amdgcn_global_load_async_to_lds_b128(
      (as1_v4i)(unsigned long long)gsrc,
      (as3_v4i)(unsigned)(unsigned long long)lds_dst,
      0, 0);
}
#endif

// ---------------------------------------------------------------------------
// Kernel 0: fp32 -> f16 convert (for x)
// ---------------------------------------------------------------------------
__global__ void convert_f16(const float* __restrict__ src,
                            _Float16* __restrict__ dst, int n) {
  int i = blockIdx.x * blockDim.x + threadIdx.x;
  if (i < n) dst[i] = (_Float16)src[i];
}

// ---------------------------------------------------------------------------
// Kernel 1: pack a row-major [K,N] fp32 weight matrix into WMMA B-fragment
// lane order (f16). Fragment f = kt*(N/16)+nt; lane l holds, contiguously,
// halves j=0..15 with K = kt*32 + (l>>4)*16 + j, N = nt*16 + (l&15).
// blockIdx.z selects the matrix (stride K*N in both src and dst).
// ---------------------------------------------------------------------------
__global__ void pack_w(const float* __restrict__ src, _Float16* __restrict__ dst,
                       int K, int N) {
  const int mat = blockIdx.z;
  const float*    s = src + (size_t)mat * K * N;
  _Float16*       d = dst + (size_t)mat * K * N;
  int idx = blockIdx.x * blockDim.x + threadIdx.x;
  if (idx >= K * N) return;
  int j    = idx & 15;
  int lane = (idx >> 4) & 31;
  int f    = idx >> 9;
  int Ntiles = N >> 4;
  int kt = f / Ntiles, nt = f % Ntiles;
  int k = kt * 32 + (lane >> 4) * 16 + j;
  int n = nt * 16 + (lane & 15);
  d[idx] = (_Float16)s[(size_t)k * N + n];
}

// ---------------------------------------------------------------------------
// Kernel 2: QKV projection. grid = (128 row-groups, 3*H). 8 waves/block,
// each wave computes a 16x64 tile: 16 K-chunks x 4 WMMAs = 64 WMMAs.
// Fully-unrolled software pipeline: chunk kc+1's fragments are in flight
// while chunk kc's WMMAs execute (partial loadcnt waits, no register copies).
// q,k stored row-major [b,h,s,64] (f16); v stored transposed [b,h,64,s].
// ---------------------------------------------------------------------------
__global__ __launch_bounds__(256) void qkv_kernel(
    const _Float16* __restrict__ xb, const _Float16* __restrict__ wpack,
    const float* __restrict__ bq, const float* __restrict__ bk,
    const float* __restrict__ bv, _Float16* __restrict__ qbf,
    _Float16* __restrict__ kbf, _Float16* __restrict__ vTb) {
  const int lane = threadIdx.x & 31, wave = threadIdx.x >> 5;
  const int hf = lane >> 4, ll = lane & 15;
  const int ph = blockIdx.y;          // 0..23 : proj*H + head
  const int p  = ph >> 3, h = ph & 7;
  const int m0 = (blockIdx.x * 8 + wave) * 16;     // row in [0,16384)
  const int b  = m0 >> 10, s0 = m0 & 1023;

  const _Float16* wp = wpack + (size_t)ph * (16 * 4 * 32 * 16);
  const v8f vzero = {0.f, 0.f, 0.f, 0.f, 0.f, 0.f, 0.f, 0.f};
  v8f acc[4] = {vzero, vzero, vzero, vzero};

  v16h a = load_a_frag(xb, m0 + ll, ND, 0, hf);
  v16h bf[4];
#pragma unroll
  for (int nt = 0; nt < 4; ++nt)
    bf[nt] = load_frag16(wp + ((size_t)nt * 32 + lane) * 16);

#pragma unroll
  for (int kc = 0; kc < 16; ++kc) {
    v16h an;
    v16h bn[4];
    if (kc < 15) {
      an = load_a_frag(xb, m0 + ll, ND, (kc + 1) * 32, hf);
#pragma unroll
      for (int nt = 0; nt < 4; ++nt)
        bn[nt] = load_frag16(wp + ((size_t)((kc + 1) * 4 + nt) * 32 + lane) * 16);
    }
#pragma unroll
    for (int nt = 0; nt < 4; ++nt) acc[nt] = wmma16(a, bf[nt], acc[nt]);
    if (kc < 15) {
      a = an;
#pragma unroll
      for (int nt = 0; nt < 4; ++nt) bf[nt] = bn[nt];
    }
  }

  const float* bias  = (p == 0) ? bq : (p == 1) ? bk : bv;
  const float* biasH = bias + h * 64;
  const int bh = b * NH + h;
#pragma unroll
  for (int nt = 0; nt < 4; ++nt) {
#pragma unroll
    for (int r = 0; r < 8; ++r) {
      const int rl = r + 8 * hf;                    // row within tile
      const int n  = nt * 16 + ll;                  // output dim
      const float val = acc[nt][r] + biasH[n];
      if (p == 2) {
        vTb[((size_t)bh * 64 + n) * NS + s0 + rl] = (_Float16)val;
      } else {
        _Float16* dst = (p == 0) ? qbf : kbf;
        dst[((size_t)bh * NS + s0 + rl) * 64 + n] = (_Float16)val;
      }
    }
  }
}

// ---------------------------------------------------------------------------
// Kernel 3: flash attention. grid = (B*H, S/128). 8 waves/block, one 16-query
// tile per wave; 32-key K/V blocks double-buffered in LDS via async
// global->LDS (ASYNCcnt), shared by all 8 waves: one barrier per key-block.
// Online softmax is lane-local (C-frag rows == stat rows per lane), 16-lane
// shfl reductions across the key dimension. P re-layout via per-wave LDS.
// ---------------------------------------------------------------------------
__global__ __launch_bounds__(256) void attn_kernel(
    const _Float16* __restrict__ qbf, const _Float16* __restrict__ kbf,
    const _Float16* __restrict__ vTb, _Float16* __restrict__ zbf) {
  __shared__ __align__(16) _Float16 kblk[2][32 * 64];    // [buf][t][dk]
  __shared__ __align__(16) _Float16 vblk[2][64 * 32];    // [buf][dv][t]
  __shared__ __align__(16) _Float16 pscr[8 * 16 * 32];   // per-wave P scratch

  const int tid = threadIdx.x;
  const int lane = tid & 31, wave = tid >> 5;
  const int hf = lane >> 4, ll = lane & 15;
  const int bh = blockIdx.x;
  const int b = bh >> 3, h = bh & 7;
  const int qs0 = blockIdx.y * 128 + wave * 16;

  const _Float16* qh = qbf + (size_t)bh * NS * 64;
  const _Float16* kh = kbf + (size_t)bh * NS * 64;
  const _Float16* vh = vTb + (size_t)bh * 64 * NS;

  const v16h aq0 = load_a_frag(qh, qs0 + ll, 64, 0, hf);
  const v16h aq1 = load_a_frag(qh, qs0 + ll, 64, 32, hf);

  const v8f vzero = {0.f, 0.f, 0.f, 0.f, 0.f, 0.f, 0.f, 0.f};
  v8f accO[4] = {vzero, vzero, vzero, vzero};
  float mrow[8], lrow[8];
#pragma unroll
  for (int r = 0; r < 8; ++r) { mrow[r] = -1e30f; lrow[r] = 0.f; }

  _Float16* pw = pscr + wave * 512;
  const int st = tid >> 3, so = (tid & 7) * 8;   // k staging: 32 rows x 128B
  const int sdv = tid >> 2, so2 = (tid & 3) * 8; // v staging: 64 rows x 64B

  auto stage = [&](int buf, int t0) {
#if defined(USE_ASYNC_LDS)
    async_copy16(&kblk[buf][st * 64 + so],   &kh[(size_t)(t0 + st) * 64 + so]);
    async_copy16(&vblk[buf][sdv * 32 + so2], &vh[(size_t)sdv * NS + t0 + so2]);
#else
    *(v8h*)&kblk[buf][st * 64 + so] =
        *(const v8h*)&kh[(size_t)(t0 + st) * 64 + so];
    *(v8h*)&vblk[buf][sdv * 32 + so2] =
        *(const v8h*)&vh[(size_t)sdv * NS + t0 + so2];
#endif
  };

  stage(0, 0);
#if defined(USE_ASYNC_LDS)
  asm volatile("s_wait_asynccnt 0" ::: "memory");
#endif
  __syncthreads();

#pragma unroll 2
  for (int kb = 0; kb < 32; ++kb) {
    // prefetch next block into the other buffer while computing this one
    if (kb + 1 < 32) stage((kb + 1) & 1, (kb + 1) * 32);

    const _Float16* kp = kblk[kb & 1];
    const _Float16* vp = vblk[kb & 1];

    // scores: S[16q x 32keys] = Q(16x64) @ K^T -> two 16x16 tiles, K-dim 64
    v8f sc0 = vzero, sc1 = vzero;
#pragma unroll
    for (int kc = 0; kc < 2; ++kc) {
      // row-major k feeds B-frags contiguously: lane = key, halves walk dk
      v16h b0 = load_frag16(&kp[(0 * 16 + ll) * 64 + kc * 32 + hf * 16]);
      v16h b1 = load_frag16(&kp[(1 * 16 + ll) * 64 + kc * 32 + hf * 16]);
      v16h a = kc ? aq1 : aq0;
      sc0 = wmma16(a, b0, sc0);
      sc1 = wmma16(a, b1, sc1);
    }

    const float scale = 0.125f;   // 1/sqrt(64)
#pragma unroll
    for (int r = 0; r < 8; ++r) {
      float x0 = sc0[r] * scale, x1 = sc1[r] * scale;
      float mx = fmaxf(x0, x1);
      mx = fmaxf(mx, __shfl_xor(mx, 1, 32));
      mx = fmaxf(mx, __shfl_xor(mx, 2, 32));
      mx = fmaxf(mx, __shfl_xor(mx, 4, 32));
      mx = fmaxf(mx, __shfl_xor(mx, 8, 32));
      float mnew = fmaxf(mrow[r], mx);
      float p0 = __expf(x0 - mnew);
      float p1 = __expf(x1 - mnew);
      float sum = p0 + p1;
      sum += __shfl_xor(sum, 1, 32);
      sum += __shfl_xor(sum, 2, 32);
      sum += __shfl_xor(sum, 4, 32);
      sum += __shfl_xor(sum, 8, 32);
      float corr = __expf(mrow[r] - mnew);
      lrow[r] = lrow[r] * corr + sum;
      mrow[r] = mnew;
      accO[0][r] *= corr; accO[1][r] *= corr;
      accO[2][r] *= corr; accO[3][r] *= corr;
      // spill P (C-frag layout) to per-wave LDS, [m][t] row-major
      const int m = r + 8 * hf;
      pw[m * 32 + ll]      = (_Float16)p0;
      pw[m * 32 + 16 + ll] = (_Float16)p1;
    }
    asm volatile("s_wait_dscnt 0" ::: "memory");  // wave-local LDS RAW

    // reload P as an A-fragment (16x32)
    v16h pa;
    {
      const _Float16* pp = pw + ll * 32 + hf * 16;
      *(v8h*)&pa       = *(const v8h*)pp;
      *((v8h*)&pa + 1) = *(const v8h*)(pp + 8);
    }
    // O += P @ V   (V transposed in LDS -> contiguous B-frags)
#pragma unroll
    for (int nt = 0; nt < 4; ++nt) {
      v16h bvf = load_frag16(&vp[(nt * 16 + ll) * 32 + hf * 16]);
      accO[nt] = wmma16(pa, bvf, accO[nt]);
    }

    // publish async data for next block + protect the buffer we overwrite next
#if defined(USE_ASYNC_LDS)
    asm volatile("s_wait_asynccnt 0" ::: "memory");
#endif
    __syncthreads();
  }

  // epilogue: O / l, concat heads into z[b, s, h*64+dv] (f16)
#pragma unroll
  for (int nt = 0; nt < 4; ++nt) {
#pragma unroll
    for (int r = 0; r < 8; ++r) {
      const float o = accO[nt][r] / lrow[r];
      const int row = qs0 + r + 8 * hf;
      const int col = h * 64 + nt * 16 + ll;
      zbf[((size_t)b * NS + row) * ND + col] = (_Float16)o;
    }
  }
}

// ---------------------------------------------------------------------------
// Kernel 4: output projection  out = z @ Wo + bo  (fp32 out).
// grid = (512/64 col-groups, 16384/128 row-groups), 8 waves/block.
// Same fully-unrolled software pipeline as the QKV GEMM.
// ---------------------------------------------------------------------------
__global__ __launch_bounds__(256) void out_gemm(
    const _Float16* __restrict__ zbf, const _Float16* __restrict__ wop,
    const float* __restrict__ bo, float* __restrict__ out) {
  const int lane = threadIdx.x & 31, wave = threadIdx.x >> 5;
  const int hf = lane >> 4, ll = lane & 15;
  const int n0 = blockIdx.x * 64;
  const int m0 = (blockIdx.y * 8 + wave) * 16;
  const v8f vzero = {0.f, 0.f, 0.f, 0.f, 0.f, 0.f, 0.f, 0.f};
  v8f acc[4] = {vzero, vzero, vzero, vzero};

  v16h a = load_a_frag(zbf, m0 + ll, ND, 0, hf);
  v16h bf[4];
#pragma unroll
  for (int nt = 0; nt < 4; ++nt)
    bf[nt] = load_frag16(wop + ((size_t)(blockIdx.x * 4 + nt) * 32 + lane) * 16);

#pragma unroll
  for (int kc = 0; kc < 16; ++kc) {
    v16h an;
    v16h bn[4];
    if (kc < 15) {
      an = load_a_frag(zbf, m0 + ll, ND, (kc + 1) * 32, hf);
#pragma unroll
      for (int nt = 0; nt < 4; ++nt) {
        const int f = (kc + 1) * 32 + (blockIdx.x * 4 + nt);  // Ntiles = 32
        bn[nt] = load_frag16(wop + ((size_t)f * 32 + lane) * 16);
      }
    }
#pragma unroll
    for (int nt = 0; nt < 4; ++nt) acc[nt] = wmma16(a, bf[nt], acc[nt]);
    if (kc < 15) {
      a = an;
#pragma unroll
      for (int nt = 0; nt < 4; ++nt) bf[nt] = bn[nt];
    }
  }
#pragma unroll
  for (int nt = 0; nt < 4; ++nt) {
#pragma unroll
    for (int r = 0; r < 8; ++r) {
      const int row = m0 + r + 8 * hf;
      const int col = n0 + nt * 16 + ll;
      out[(size_t)row * ND + col] = acc[nt][r] + bo[col];
    }
  }
}

// ---------------------------------------------------------------------------
extern "C" void kernel_launch(void* const* d_in, const int* in_sizes, int n_in,
                              void* d_out, int out_size, void* d_ws,
                              size_t ws_size, hipStream_t stream) {
  (void)in_sizes; (void)n_in; (void)out_size; (void)ws_size;
  const float* x  = (const float*)d_in[0];
  const float* Wq = (const float*)d_in[1];
  const float* bq = (const float*)d_in[2];
  const float* Wk = (const float*)d_in[3];
  const float* bk = (const float*)d_in[4];
  const float* Wv = (const float*)d_in[5];
  const float* bv = (const float*)d_in[6];
  const float* Wo = (const float*)d_in[7];
  const float* bo = (const float*)d_in[8];
  float* out = (float*)d_out;

  char* ws = (char*)d_ws;
  const size_t nBS_D = (size_t)NB * NS * ND;        // 8,388,608
  const size_t nQKV  = (size_t)NB * NH * NS * NDK;  // 8,388,608
  _Float16* xb   = (_Float16*)ws; ws += nBS_D * 2;
  _Float16* wqkv = (_Float16*)ws; ws += (size_t)24 * ND * NDK * 2;
  _Float16* wop  = (_Float16*)ws; ws += (size_t)ND * ND * 2;
  _Float16* qbf  = (_Float16*)ws; ws += nQKV * 2;
  _Float16* kbf  = (_Float16*)ws; ws += nQKV * 2;
  _Float16* vTb  = (_Float16*)ws; ws += nQKV * 2;
  _Float16* zbf  = (_Float16*)ws; ws += nBS_D * 2;

  // 0) convert x to f16
  convert_f16<<<(int)(nBS_D / 256), 256, 0, stream>>>(x, xb, (int)nBS_D);

  // 1) pack weights into WMMA B-fragment order
  const size_t matQKV = (size_t)ND * NDK;           // 32768 halves per head
  pack_w<<<dim3(128, 1, NH), 256, 0, stream>>>(Wq, wqkv + 0 * NH * matQKV, ND, NDK);
  pack_w<<<dim3(128, 1, NH), 256, 0, stream>>>(Wk, wqkv + 1 * NH * matQKV, ND, NDK);
  pack_w<<<dim3(128, 1, NH), 256, 0, stream>>>(Wv, wqkv + 2 * NH * matQKV, ND, NDK);
  pack_w<<<dim3(1024, 1, 1), 256, 0, stream>>>(Wo, wop, ND, ND);

  // 2) QKV projections
  qkv_kernel<<<dim3(128, 24), 256, 0, stream>>>(xb, wqkv, bq, bk, bv, qbf, kbf, vTb);

  // 3) flash attention
  attn_kernel<<<dim3(NB * NH, NS / 128), 256, 0, stream>>>(qbf, kbf, vTb, zbf);

  // 4) output projection
  out_gemm<<<dim3(ND / 64, 128), 256, 0, stream>>>(zbf, wop, bo, out);
}
